// Qwen3MoeForCausalLM_49443663511919
// MI455X (gfx1250) — matrix operand order repaired and checked by reading, not compiled, not run on hardware
//
#include <hip/hip_runtime.h>
#include <hip/hip_bf16.h>

#define T_TOK 1024
#define HDIM  2048
#define IDIM  768
#define NEXP  8

typedef __attribute__((ext_vector_type(16))) __bf16 v16bf;
typedef __attribute__((ext_vector_type(8)))  float  v8f;

union FragBF { uint4 q[2]; v16bf v; };
union Pack4BF { __bf16 h[4]; unsigned long long u64; };

// A-matrix fragment, 16x32 bf16 (MxK), contiguous K order, same as load_fragB:
// lanes 0-15: row M=lane, K = 0..15; lanes 16-31: row M=lane-16, K = 16..31
__device__ inline v16bf load_fragA(const __bf16* __restrict__ s, int lane) {
    const int m   = lane & 15;
    const int grp = lane >> 4;
    const int k0  = grp ? 16 : 0;
    const int k1  = k0 + 8;
    FragBF f;
    f.q[0] = *reinterpret_cast<const uint4*>(s + m * 32 + k0);
    f.q[1] = *reinterpret_cast<const uint4*>(s + m * 32 + k1);
    return f.v;
}

// B-matrix fragment, 32x16 bf16 (KxN), stored as W rows (N-major, K contiguous):
// lanes 0-15: col N=lane, K = 0..15; lanes 16-31: col N=lane-16, K = 16..31
__device__ inline v16bf load_fragB(const __bf16* __restrict__ s, int lane, int ct) {
    const int n = ct * 16 + (lane & 15);
    const int k = (lane >> 4) ? 16 : 0;
    FragBF f;
    f.q[0] = *reinterpret_cast<const uint4*>(s + n * 32 + k);
    f.q[1] = *reinterpret_cast<const uint4*>(s + n * 32 + k + 8);
    return f.v;
}

// Async-copy one 128x32 bf16 tile (8 KB) from global to LDS.
// 512 x 16-byte chunks, 2 per thread. ASYNCcnt-tracked.
__device__ inline void async_tileA(const __bf16* __restrict__ src,  // origin + r0*ld
                                   const __bf16* lds_tile, int tid, int ld, int kk) {
#pragma unroll
    for (int i = 0; i < 2; ++i) {
        int c   = tid + 256 * i;          // chunk id
        int row = c >> 2;                 // 4 chunks per 32-elem row
        int col = (c & 3) * 8;
        const __bf16* g = src + (size_t)row * ld + kk + col;
        // Low 32 bits of a flat LDS address are the wave-relative LDS byte offset (ISA 10.2)
        unsigned lds = (unsigned)(unsigned long long)(lds_tile + c * 8);
        asm volatile("global_load_async_to_lds_b128 %0, %1, off"
                     :: "v"(lds), "v"((unsigned long long)g)
                     : "memory");
    }
}

__device__ inline void wait_async_lds() {
    asm volatile("s_wait_asynccnt 0" ::: "memory");
}

__device__ inline unsigned long long pack4(float4 v) {
    Pack4BF p;
    p.h[0] = (__bf16)v.x; p.h[1] = (__bf16)v.y;
    p.h[2] = (__bf16)v.z; p.h[3] = (__bf16)v.w;
    return p.u64;
}

__global__ void zero_out_kernel(float4* __restrict__ out, int n4) {
    int i = blockIdx.x * 256 + threadIdx.x;
    if (i < n4) out[i] = make_float4(0.f, 0.f, 0.f, 0.f);
}

__global__ void cvt_hs_kernel(const float4* __restrict__ hs,
                              unsigned long long* __restrict__ hs16, int n4) {
    int i = blockIdx.x * 256 + threadIdx.x;
    if (i < n4) hs16[i] = pack4(hs[i]);
}

// Router: fp32 logits -> softmax -> top2 -> renormalized combine weights comb[T,E]
__global__ void moe_router(const float* __restrict__ hs,
                           const float* __restrict__ gw,
                           float* __restrict__ comb) {
    int t = blockIdx.x * blockDim.x + threadIdx.x;
    if (t >= T_TOK) return;
    float lg[NEXP];
#pragma unroll
    for (int e = 0; e < NEXP; ++e) lg[e] = 0.0f;
    for (int h = 0; h < HDIM; ++h) {
        float x = hs[(size_t)t * HDIM + h];
#pragma unroll
        for (int e = 0; e < NEXP; ++e)
            lg[e] = fmaf(x, gw[e * HDIM + h], lg[e]);
    }
    float m = lg[0];
#pragma unroll
    for (int e = 1; e < NEXP; ++e) m = fmaxf(m, lg[e]);
    float p[NEXP]; float s = 0.0f;
#pragma unroll
    for (int e = 0; e < NEXP; ++e) { p[e] = __expf(lg[e] - m); s += p[e]; }
    float inv_s = 1.0f / s;
#pragma unroll
    for (int e = 0; e < NEXP; ++e) p[e] *= inv_s;
    int i1 = 0;
#pragma unroll
    for (int e = 1; e < NEXP; ++e) if (p[e] > p[i1]) i1 = e;
    int i2 = (i1 == 0) ? 1 : 0;
#pragma unroll
    for (int e = 0; e < NEXP; ++e) if (e != i1 && p[e] > p[i2]) i2 = e;
    float inv_tk = 1.0f / (p[i1] + p[i2]);
#pragma unroll
    for (int e = 0; e < NEXP; ++e) {
        float w = (e == i1) ? p[i1] * inv_tk : ((e == i2) ? p[i2] * inv_tk : 0.0f);
        comb[t * NEXP + e] = w;
    }
}

// Expert gate+up proj with fused SiLU*mul, double-buffered pipeline.
// Block: 128 tokens x 64 intermediate cols (gate+up halves together).
// hs16: bf16 [T,H]; wgu: expert slice [2I,H] f32. Writes bf16 h[T,I].
__global__ __launch_bounds__(256)
void moe_gateup_silu(const __bf16* __restrict__ hs16,
                     const float* __restrict__ wgu,
                     __bf16* __restrict__ hsc) {
    __shared__ __bf16 sA [2][128 * 32];
    __shared__ __bf16 sBg[2][64 * 32];
    __shared__ __bf16 sBu[2][64 * 32];

    const int tid  = threadIdx.x;
    const int lane = tid & 31;
    const int wave = tid >> 5;
    const int r0   = blockIdx.x * 128;
    const int n0   = blockIdx.y * 64;
    const __bf16* Abase = hs16 + (size_t)r0 * HDIM;

    v8f accg[4], accu[4];
    v8f zf = {};
#pragma unroll
    for (int i = 0; i < 4; ++i) { accg[i] = zf; accu[i] = zf; }

    const int NT = HDIM / 32;

    // B-tile staging geometry: 64 rows x 8 float4/row = 512 float4, 2 per thread.
    // chunk c: row = c>>3, q = c&7 (float4 within the 32-col row)

    // ---- prologue: fill buffer 0 ----
    async_tileA(Abase, sA[0], tid, HDIM, 0);
    {
        float4 bg[2], bu[2];
#pragma unroll
        for (int i = 0; i < 2; ++i) {
            int c = tid + 256 * i, row = c >> 3, q = c & 7;
            bg[i] = *reinterpret_cast<const float4*>(&wgu[(size_t)(n0 + row) * HDIM + q * 4]);
            bu[i] = *reinterpret_cast<const float4*>(&wgu[(size_t)(IDIM + n0 + row) * HDIM + q * 4]);
        }
#pragma unroll
        for (int i = 0; i < 2; ++i) {
            int c = tid + 256 * i, row = c >> 3, q = c & 7;
            *reinterpret_cast<unsigned long long*>(&sBg[0][row * 32 + q * 4]) = pack4(bg[i]);
            *reinterpret_cast<unsigned long long*>(&sBu[0][row * 32 + q * 4]) = pack4(bu[i]);
        }
    }
    wait_async_lds();
    __syncthreads();

    // ---- main pipelined loop ----
    for (int kt = 0; kt < NT; ++kt) {
        const int cur = kt & 1;
        const int nxt = cur ^ 1;
        const bool more = (kt + 1 < NT);
        const int kk1 = (kt + 1) * 32;

        float4 bg[2], bu[2];
        if (more) {
            async_tileA(Abase, sA[nxt], tid, HDIM, kk1);
#pragma unroll
            for (int i = 0; i < 2; ++i) {
                int c = tid + 256 * i, row = c >> 3, q = c & 7;
                bg[i] = *reinterpret_cast<const float4*>(&wgu[(size_t)(n0 + row) * HDIM + kk1 + q * 4]);
                bu[i] = *reinterpret_cast<const float4*>(&wgu[(size_t)(IDIM + n0 + row) * HDIM + kk1 + q * 4]);
            }
        }

        v16bf a = load_fragA(sA[cur] + wave * 16 * 32, lane);
#pragma unroll
        for (int ct = 0; ct < 4; ++ct) {
            v16bf bgf = load_fragB(sBg[cur], lane, ct);
            accg[ct] = __builtin_amdgcn_wmma_f32_16x16x32_bf16(
                false, a, false, bgf, (short)0, accg[ct], false, false);
            v16bf buf = load_fragB(sBu[cur], lane, ct);
            accu[ct] = __builtin_amdgcn_wmma_f32_16x16x32_bf16(
                false, a, false, buf, (short)0, accu[ct], false, false);
        }

        if (more) {
#pragma unroll
            for (int i = 0; i < 2; ++i) {
                int c = tid + 256 * i, row = c >> 3, q = c & 7;
                *reinterpret_cast<unsigned long long*>(&sBg[nxt][row * 32 + q * 4]) = pack4(bg[i]);
                *reinterpret_cast<unsigned long long*>(&sBu[nxt][row * 32 + q * 4]) = pack4(bu[i]);
            }
        }
        wait_async_lds();
        __syncthreads();
    }

    // ---- epilogue: h = silu(g) * u, store bf16 ----
    const int grp = lane >> 4, nl = lane & 15;
#pragma unroll
    for (int ct = 0; ct < 4; ++ct) {
        int col = n0 + ct * 16 + nl;
#pragma unroll
        for (int r = 0; r < 8; ++r) {
            int row = r0 + wave * 16 + r + 8 * grp;
            float g = accg[ct][r];
            float u = accu[ct][r];
            float h = (g / (1.0f + __expf(-g))) * u;
            hsc[(size_t)row * IDIM + col] = (__bf16)h;
        }
    }
}

// Expert down proj with routed-weight accumulation: out[t,n] += comb[t,e] * (h @ Wd^T)
// Block: 128 tokens x 128 hidden cols, double-buffered.
__global__ __launch_bounds__(256)
void moe_down_acc(const __bf16* __restrict__ hsc,
                  const float* __restrict__ wd,
                  const float* __restrict__ comb,
                  float* __restrict__ out, int e) {
    __shared__ __bf16 sA[2][128 * 32];
    __shared__ __bf16 sB[2][128 * 32];

    const int tid  = threadIdx.x;
    const int lane = tid & 31;
    const int wave = tid >> 5;
    const int r0   = blockIdx.x * 128;
    const int c0   = blockIdx.y * 128;
    const __bf16* Abase = hsc + (size_t)r0 * IDIM;

    v8f acc[8];
    v8f zf = {};
#pragma unroll
    for (int i = 0; i < 8; ++i) acc[i] = zf;

    const int NT = IDIM / 32;

    // B-tile staging: 128 rows x 8 float4/row = 1024 float4, 4 per thread.

    // ---- prologue ----
    async_tileA(Abase, sA[0], tid, IDIM, 0);
    {
        float4 b[4];
#pragma unroll
        for (int i = 0; i < 4; ++i) {
            int c = tid + 256 * i, row = c >> 3, q = c & 7;
            b[i] = *reinterpret_cast<const float4*>(&wd[(size_t)(c0 + row) * IDIM + q * 4]);
        }
#pragma unroll
        for (int i = 0; i < 4; ++i) {
            int c = tid + 256 * i, row = c >> 3, q = c & 7;
            *reinterpret_cast<unsigned long long*>(&sB[0][row * 32 + q * 4]) = pack4(b[i]);
        }
    }
    wait_async_lds();
    __syncthreads();

    // ---- main pipelined loop ----
    for (int kt = 0; kt < NT; ++kt) {
        const int cur = kt & 1;
        const int nxt = cur ^ 1;
        const bool more = (kt + 1 < NT);
        const int kk1 = (kt + 1) * 32;

        float4 b[4];
        if (more) {
            async_tileA(Abase, sA[nxt], tid, IDIM, kk1);
#pragma unroll
            for (int i = 0; i < 4; ++i) {
                int c = tid + 256 * i, row = c >> 3, q = c & 7;
                b[i] = *reinterpret_cast<const float4*>(&wd[(size_t)(c0 + row) * IDIM + kk1 + q * 4]);
            }
        }

        v16bf a = load_fragA(sA[cur] + wave * 16 * 32, lane);
#pragma unroll
        for (int ct = 0; ct < 8; ++ct) {
            v16bf bf = load_fragB(sB[cur], lane, ct);
            acc[ct] = __builtin_amdgcn_wmma_f32_16x16x32_bf16(
                false, a, false, bf, (short)0, acc[ct], false, false);
        }

        if (more) {
#pragma unroll
            for (int i = 0; i < 4; ++i) {
                int c = tid + 256 * i, row = c >> 3, q = c & 7;
                *reinterpret_cast<unsigned long long*>(&sB[nxt][row * 32 + q * 4]) = pack4(b[i]);
            }
        }
        wait_async_lds();
        __syncthreads();
    }

    // ---- epilogue: out += comb * acc (unique (row,col) per lane-element; launches serialized) ----
    const int grp = lane >> 4, nl = lane & 15;
    float wgt[8];
#pragma unroll
    for (int r = 0; r < 8; ++r)
        wgt[r] = comb[(r0 + wave * 16 + r + 8 * grp) * NEXP + e];
#pragma unroll
    for (int ct = 0; ct < 8; ++ct) {
        int col = c0 + ct * 16 + nl;
#pragma unroll
        for (int r = 0; r < 8; ++r) {
            int row = r0 + wave * 16 + r + 8 * grp;
            size_t o = (size_t)row * HDIM + col;
            out[o] += wgt[r] * acc[ct][r];
        }
    }
}

extern "C" void kernel_launch(void* const* d_in, const int* in_sizes, int n_in,
                              void* d_out, int out_size, void* d_ws, size_t ws_size,
                              hipStream_t stream) {
    const float* hs  = (const float*)d_in[0];  // [T, H]
    const float* gw  = (const float*)d_in[1];  // [E, H]
    const float* wgu = (const float*)d_in[2];  // [E, 2I, H]
    const float* wd  = (const float*)d_in[3];  // [E, H, I]
    float* out = (float*)d_out;                // [T, H]

    // workspace: comb[T,E] f32 (32 KB) | h[T,I] bf16 (1.5 MB) | hs16[T,H] bf16 (4 MB)
    float*  comb = (float*)d_ws;
    __bf16* hsc  = (__bf16*)((char*)d_ws + (size_t)T_TOK * NEXP * sizeof(float));
    __bf16* hs16 = hsc + (size_t)T_TOK * IDIM;

    const int n4 = T_TOK * HDIM / 4;
    zero_out_kernel<<<(n4 + 255) / 256, 256, 0, stream>>>((float4*)out, n4);
    cvt_hs_kernel<<<(n4 + 255) / 256, 256, 0, stream>>>(
        (const float4*)hs, (unsigned long long*)hs16, n4);
    moe_router<<<T_TOK / 256, 256, 0, stream>>>(hs, gw, comb);

    for (int e = 0; e < NEXP; ++e) {
        const float* wgu_e = wgu + (size_t)e * 2 * IDIM * HDIM;
        const float* wd_e  = wd  + (size_t)e * HDIM * IDIM;
        moe_gateup_silu<<<dim3(T_TOK / 128, IDIM / 64), 256, 0, stream>>>(hs16, wgu_e, hsc);
        moe_down_acc<<<dim3(T_TOK / 128, HDIM / 128), 256, 0, stream>>>(hsc, wd_e, comb, out, e);
    }
}
